// GRURNN_49349174231072
// MI455X (gfx1250) — matrix operand order repaired and checked by reading, not compiled, not run on hardware
//
#include <hip/hip_runtime.h>
#include <math.h>

// ---------------------------------------------------------------------------
// Conv-GRU for MI455X (gfx1250, wave32, WMMA).
//
// GEMM mapping per 16x16-pixel tile, per output row of 16 pixels:
//   D(16x16) = A(16x32) x B(32x16) + C   via v_wmma_f32_16x16x32_f16
//   M = packed output channels (z|r pack, or h / head channels)
//   N = 16 consecutive pixels (-> coalesced f32 stores from D VGPRs)
//   K = tap*16 + ci ; 9 taps * 16 ch slots = 144, padded to 160 = 5 WMMAs
// Dilation and channel counts are template parameters so all LDS addressing
// constant-folds; epilogue activations use hardware v_exp_f32/v_rcp_f32.
// ---------------------------------------------------------------------------

typedef __attribute__((ext_vector_type(16))) _Float16 v16h;
typedef __attribute__((ext_vector_type(8)))  float    v8f;

#define IMG_H 128
#define IMG_W 128
#define IMG_HW (IMG_H * IMG_W)
#define NB 8          // batch
#define HN 8          // hidden channels
#define NT 20         // timesteps
#define KSTEPS 5      // ceil(144/32)
#define WF_STRIDE (KSTEPS * 32 * 16)   // halves per prepped conv (2560)

// ---------------------------------------------------------------------------
// Weight prep: build the exact per-lane A-fragment stream for WMMA f16 A.
// 16-bit A 16x32 layout used here (contiguous k, matched by B): lane row m = lane%16;
//   lanes 0-15 : halves j=0..15 -> K = s*32 + j
//   lanes 16-31: halves j=0..15 -> K = s*32 + 16 + j
// K decodes as tap = K/16, ci = K%16. Rows 0..7 from wA, rows 8..15 from wB
// (z|r packing); missing -> 0.
// ---------------------------------------------------------------------------
__global__ void prep_wfrag(const float* __restrict__ wA, const float* __restrict__ wB,
                           int Cin, int CoutA, int CoutB, _Float16* __restrict__ dst)
{
    int idx = blockIdx.x * blockDim.x + threadIdx.x;
    if (idx >= WF_STRIDE) return;
    int j    = idx & 15;
    int lane = (idx >> 4) & 31;
    int s    = idx >> 9;
    int m     = lane & 15;
    int kbase = (lane < 16) ? 0 : 16;
    int K     = s * 32 + kbase + j;
    int tap = K >> 4;     // 0..9 (tap 9 = pad)
    int ci  = K & 15;
    float v = 0.f;
    if (tap < 9 && ci < Cin) {
        if (m < CoutA && wA)                      v = wA[(m * Cin + ci) * 9 + tap];
        else if (m >= 8 && (m - 8) < CoutB && wB) v = wB[((m - 8) * Cin + ci) * 9 + tap];
    }
    dst[idx] = (_Float16)v;
}

// ---------------------------------------------------------------------------
// Fast activations on hardware transcendentals (v_exp_f32 is exp2).
// ---------------------------------------------------------------------------
#define LOG2E 1.4426950408889634f
__device__ __forceinline__ float fsigmoid(float x)
{
    return __builtin_amdgcn_rcpf(1.f + __builtin_amdgcn_exp2f(-LOG2E * x));
}
__device__ __forceinline__ float ftanh(float x)
{
    return 1.f - 2.f * __builtin_amdgcn_rcpf(1.f + __builtin_amdgcn_exp2f(2.f * LOG2E * x));
}

// ---------------------------------------------------------------------------
// Stage a halo-extended 16x16 tile into LDS as [y][x][16ch] f16.
// Channel map: slots [0..N0) = p0 (optionally multiplied by m0, for r*h),
//              slots [N0..N0+N1) = p1, rest zero.  null plane -> zeros.
// ---------------------------------------------------------------------------
template <int WL, int HALO, int N0, int N1>
__device__ __forceinline__ void stage_tile(_Float16* smem, int gx0, int gy0,
                                           const float* p0, const float* m0,
                                           const float* p1)
{
    constexpr int NPIX = WL * WL;
    for (int i = threadIdx.x; i < NPIX; i += blockDim.x) {
        int py = i / WL, px = i - py * WL;       // WL is constexpr -> cheap
        int gy = gy0 + py - HALO, gx = gx0 + px - HALO;
        union { _Float16 h[16]; v16h v; } u;
        u.v = (v16h){};
        if (gy >= 0 && gy < IMG_H && gx >= 0 && gx < IMG_W) {
            int off = gy * IMG_W + gx;
            if (p0) {
#pragma unroll
                for (int c = 0; c < N0; ++c) {
                    float f = p0[(size_t)c * IMG_HW + off];
                    if (m0) f *= m0[(size_t)c * IMG_HW + off];
                    u.h[c] = (_Float16)f;
                }
            }
            if (p1) {
#pragma unroll
                for (int c = 0; c < N1; ++c)
                    u.h[N0 + c] = (_Float16)p1[(size_t)c * IMG_HW + off];
            }
        }
        *(v16h*)&smem[(size_t)i * 16] = u.v;
    }
}

// ---------------------------------------------------------------------------
// One D tile (16 channels x 16 pixels) for output row y_row.
// B fragment per lane: pixel n = lane%16 at tap (2s + lane/16): the 16
// contiguous channel halves of that LDS pixel (two ds_load_b128).
// ---------------------------------------------------------------------------
template <int WL, int HALO, int DIL>
__device__ __forceinline__ v8f conv_row(const _Float16* smem, const v16h* afrag,
                                        int y_row, int lane)
{
    v8f acc = {};
    int n  = lane & 15;
    int hi = lane >> 4;
#pragma unroll
    for (int s = 0; s < KSTEPS; ++s) {
        int tap = 2 * s + hi;
        v16h b = {};
        if (tap < 9) {
            int ky = tap / 3 - 1, kx = tap % 3 - 1;
            int yy = y_row + HALO + ky * DIL;
            int xx = n + HALO + kx * DIL;
            b = *(const v16h*)&smem[((size_t)yy * WL + xx) * 16];
        }
        acc = __builtin_amdgcn_wmma_f32_16x16x32_f16(
                false, afrag[s], false, b, (short)0, acc, false, false);
    }
    return acc;
}

__device__ __forceinline__ void load_afrag(v16h* afrag, const _Float16* wfrag, int lane)
{
#pragma unroll
    for (int s = 0; s < KSTEPS; ++s)
        afrag[s] = *(const v16h*)&wfrag[((size_t)s * 32 + lane) * 16];
}

// ---------------------------------------------------------------------------
// z & r gates fused: M rows 0-7 = z channels, 8-15 = r channels (same input).
// ---------------------------------------------------------------------------
template <int DIL, int CX>
__global__ __launch_bounds__(128)
void conv_zr(const float* __restrict__ x, const float* __restrict__ hprev,
             const _Float16* __restrict__ wfrag,
             const float* __restrict__ bz, const float* __restrict__ br,
             float* __restrict__ wsz, float* __restrict__ wsr)
{
    constexpr int WL = 16 + 2 * DIL;
    __shared__ _Float16 smem[WL * WL * 16];
    int b   = blockIdx.y;
    int gx0 = (blockIdx.x & 7) * 16, gy0 = (blockIdx.x >> 3) * 16;
    const float* xb = x + (size_t)b * CX * IMG_HW;
    const float* hb = hprev ? hprev + (size_t)b * HN * IMG_HW : nullptr;
    stage_tile<WL, DIL, CX, HN>(smem, gx0, gy0, xb, nullptr, hb);
    __syncthreads();

    int lane = threadIdx.x & 31, wave = threadIdx.x >> 5;
    v16h afrag[KSTEPS];
    load_afrag(afrag, wfrag, lane);

    int chbase = (lane >> 4) * 8;
    for (int r = 0; r < 4; ++r) {
        int y_row = wave * 4 + r;
        v8f acc = conv_row<WL, DIL, DIL>(smem, afrag, y_row, lane);
        size_t pix = (size_t)(gy0 + y_row) * IMG_W + gx0 + (lane & 15);
#pragma unroll
        for (int v = 0; v < 8; ++v) {
            int ch = chbase + v;
            float bias = (ch < 8) ? bz[ch] : br[ch - 8];
            float val  = fsigmoid(acc[v] + bias);
            if (ch < 8) wsz[((size_t)b * 8 + ch)     * IMG_HW + pix] = val;
            else        wsr[((size_t)b * 8 + ch - 8) * IMG_HW + pix] = val;
        }
    }
}

// ---------------------------------------------------------------------------
// h-gate conv on cat(r*h, x) + fused GRU update; writes h_new into hs[t][l].
// ---------------------------------------------------------------------------
template <int DIL, int CX>
__global__ __launch_bounds__(128)
void conv_h(const float* __restrict__ x, const float* __restrict__ hprev,
            const float* __restrict__ wsr, const float* __restrict__ wsz,
            const _Float16* __restrict__ wfrag, const float* __restrict__ bh,
            float* __restrict__ hout)
{
    constexpr int WL = 16 + 2 * DIL;
    __shared__ _Float16 smem[WL * WL * 16];
    int b   = blockIdx.y;
    int gx0 = (blockIdx.x & 7) * 16, gy0 = (blockIdx.x >> 3) * 16;
    const float* xb = x + (size_t)b * CX * IMG_HW;
    const float* hb = hprev ? hprev + (size_t)b * HN * IMG_HW : nullptr;
    const float* rb = wsr + (size_t)b * HN * IMG_HW;
    stage_tile<WL, DIL, HN, CX>(smem, gx0, gy0, hb, rb, xb);   // [r*h, x]
    __syncthreads();

    int lane = threadIdx.x & 31, wave = threadIdx.x >> 5;
    v16h afrag[KSTEPS];
    load_afrag(afrag, wfrag, lane);

    for (int r = 0; r < 4; ++r) {
        int y_row = wave * 4 + r;
        v8f acc = conv_row<WL, DIL, DIL>(smem, afrag, y_row, lane);
        size_t pix = (size_t)(gy0 + y_row) * IMG_W + gx0 + (lane & 15);
        if (lane < 16) {  // valid D rows are channels 0..7
#pragma unroll
            for (int v = 0; v < 8; ++v) {
                float h1 = ftanh(acc[v] + bh[v]);
                float zv = wsz[((size_t)b * 8 + v) * IMG_HW + pix];
                float hp = hb ? hb[(size_t)v * IMG_HW + pix] : 0.f;
                hout[((size_t)b * HN + v) * IMG_HW + pix] = (1.f - zv) * hp + zv * h1;
            }
        }
    }
}

// ---------------------------------------------------------------------------
// Head convs: ACT 1 = relu (COUT 8), ACT 2 = sigmoid (COUT 1).
// ---------------------------------------------------------------------------
template <int DIL, int ACT, int COUT>
__global__ __launch_bounds__(128)
void conv_head(const float* __restrict__ in, const _Float16* __restrict__ wfrag,
               const float* __restrict__ bias, float* __restrict__ out)
{
    constexpr int WL = 16 + 2 * DIL;
    __shared__ _Float16 smem[WL * WL * 16];
    int b   = blockIdx.y;
    int gx0 = (blockIdx.x & 7) * 16, gy0 = (blockIdx.x >> 3) * 16;
    stage_tile<WL, DIL, 8, 0>(smem, gx0, gy0, in + (size_t)b * 8 * IMG_HW,
                              nullptr, nullptr);
    __syncthreads();

    int lane = threadIdx.x & 31, wave = threadIdx.x >> 5;
    v16h afrag[KSTEPS];
    load_afrag(afrag, wfrag, lane);

    int chbase = (lane >> 4) * 8;
    for (int r = 0; r < 4; ++r) {
        int y_row = wave * 4 + r;
        v8f acc = conv_row<WL, DIL, DIL>(smem, afrag, y_row, lane);
        size_t pix = (size_t)(gy0 + y_row) * IMG_W + gx0 + (lane & 15);
#pragma unroll
        for (int v = 0; v < 8; ++v) {
            int ch = chbase + v;
            if (ch < COUT) {
                float val = acc[v] + bias[ch];
                val = (ACT == 1) ? fmaxf(val, 0.f) : fsigmoid(val);
                out[((size_t)b * COUT + ch) * IMG_HW + pix] = val;
            }
        }
    }
}

// hs[t][3] stays the all-zero initial state for every t.
__global__ void fill_h3(float* __restrict__ hs)
{
    const size_t per = (size_t)NB * HN * IMG_HW;
    const size_t total = (size_t)NT * per;
    size_t stride = (size_t)gridDim.x * blockDim.x;
    for (size_t i = (size_t)blockIdx.x * blockDim.x + threadIdx.x; i < total; i += stride) {
        size_t t = i / per, rph = i - t * per;
        hs[(t * 4 + 3) * per + rph] = 0.f;
    }
}

// ---------------------------------------------------------------------------
extern "C" void kernel_launch(void* const* d_in, const int* in_sizes, int n_in,
                              void* d_out, int out_size, void* d_ws, size_t ws_size,
                              hipStream_t stream)
{
    const float* X   = (const float*)d_in[0];
    const float* w1z = (const float*)d_in[1];  const float* b1z = (const float*)d_in[2];
    const float* w1r = (const float*)d_in[3];  const float* b1r = (const float*)d_in[4];
    const float* w1h = (const float*)d_in[5];  const float* b1h = (const float*)d_in[6];
    const float* w2z = (const float*)d_in[7];  const float* b2z = (const float*)d_in[8];
    const float* w2r = (const float*)d_in[9];  const float* b2r = (const float*)d_in[10];
    const float* w2h = (const float*)d_in[11]; const float* b2h = (const float*)d_in[12];
    const float* w3z = (const float*)d_in[13]; const float* b3z = (const float*)d_in[14];
    const float* w3r = (const float*)d_in[15]; const float* b3r = (const float*)d_in[16];
    const float* w3h = (const float*)d_in[17]; const float* b3h = (const float*)d_in[18];
    const float* wo1 = (const float*)d_in[19]; const float* bo1 = (const float*)d_in[20];
    const float* wo2 = (const float*)d_in[21]; const float* bo2 = (const float*)d_in[22];
    const float* wo  = (const float*)d_in[23]; const float* bo  = (const float*)d_in[24];

    float* outp = (float*)d_out;
    const size_t Y_SZ = (size_t)NT * NB * IMG_HW;           // y: [T,B,1,H,W]
    float* y  = outp;
    float* hs = outp + Y_SZ;                                 // hs: [T,4,B,HN,H,W]

    char*     ws    = (char*)d_ws;
    _Float16* wfrag = (_Float16*)ws;                         // 9 * 2560 halves
    float* wsz = (float*)(ws + 65536);
    float* wsr = wsz + (size_t)NB * 8 * IMG_HW;
    float* wy1 = wsr + (size_t)NB * 8 * IMG_HW;
    float* wy2 = wy1 + (size_t)NB * 8 * IMG_HW;

    // Pre-swizzle weights into WMMA A-fragment layout (f16).
    struct { const float *a, *b; int cin, ca, cb; } pw[9] = {
        {w1z, w1r, 9, 8, 8}, {w1h, nullptr, 9, 8, 0},
        {w2z, w2r, 16, 8, 8}, {w2h, nullptr, 16, 8, 0},
        {w3z, w3r, 16, 8, 8}, {w3h, nullptr, 16, 8, 0},
        {wo1, nullptr, 8, 8, 0}, {wo2, nullptr, 8, 8, 0}, {wo, nullptr, 8, 1, 0}};
    for (int i = 0; i < 9; ++i)
        prep_wfrag<<<10, 256, 0, stream>>>(pw[i].a, pw[i].b, pw[i].cin, pw[i].ca,
                                           pw[i].cb, wfrag + (size_t)i * WF_STRIDE);

    fill_h3<<<2048, 256, 0, stream>>>(hs);

    dim3 grid(64, NB), blk(128);
    auto hslot = [&](int t, int l) -> float* {
        return hs + (((size_t)t * 4 + l) * NB) * HN * IMG_HW;
    };

    for (int t = 0; t < NT; ++t) {
        const float* xt  = X + (size_t)t * NB * IMG_HW;
        const float* h0p = t ? hslot(t - 1, 0) : nullptr;
        const float* h1p = t ? hslot(t - 1, 1) : nullptr;
        const float* h2p = t ? hslot(t - 1, 2) : nullptr;

        // layer 1 (dil 1, Cx 1)
        conv_zr<1, 1><<<grid, blk, 0, stream>>>(xt, h0p, wfrag + 0 * WF_STRIDE, b1z, b1r, wsz, wsr);
        conv_h <1, 1><<<grid, blk, 0, stream>>>(xt, h0p, wsr, wsz, wfrag + 1 * WF_STRIDE, b1h, hslot(t, 0));
        // layer 2 (dil 2, Cx 8)
        conv_zr<2, 8><<<grid, blk, 0, stream>>>(hslot(t, 0), h1p, wfrag + 2 * WF_STRIDE, b2z, b2r, wsz, wsr);
        conv_h <2, 8><<<grid, blk, 0, stream>>>(hslot(t, 0), h1p, wsr, wsz, wfrag + 3 * WF_STRIDE, b2h, hslot(t, 1));
        // layer 3 (dil 4, Cx 8)
        conv_zr<4, 8><<<grid, blk, 0, stream>>>(hslot(t, 1), h2p, wfrag + 4 * WF_STRIDE, b3z, b3r, wsz, wsr);
        conv_h <4, 8><<<grid, blk, 0, stream>>>(hslot(t, 1), h2p, wsr, wsz, wfrag + 5 * WF_STRIDE, b3h, hslot(t, 2));
        // output head
        conv_head<2, 1, 8><<<grid, blk, 0, stream>>>(hslot(t, 2), wfrag + 6 * WF_STRIDE, bo1, wy1);
        conv_head<1, 1, 8><<<grid, blk, 0, stream>>>(wy1, wfrag + 7 * WF_STRIDE, bo2, wy2);
        conv_head<1, 2, 1><<<grid, blk, 0, stream>>>(wy2, wfrag + 8 * WF_STRIDE, bo,
                                                     y + (size_t)t * NB * IMG_HW);
    }
}
